// HeterogeneityGAT_53669911331381
// MI455X (gfx1250) — compile-verified, matrix operand order checked by reading, not run on hardware
//
#include <hip/hip_runtime.h>
#include <hip/hip_bf16.h>
#include <math.h>
#include <cstdint>

// ---------------- problem constants ----------------
#define N_NODES  60000
#define M_PAD    60032   // 469 * 128, padded row count for 128-row GEMM block tiles
#define E_EDGES  600000
#define E2       (E_EDGES + N_NODES)   // edges + self loops = 660000
#define IN_F     387
#define HID      256
#define HEADS    4
#define DH       64
#define HET      3
#define AH       128
#define B_GRAPHS 8
#define KIN_PAD  416    // 387 padded to 13*32
#define KG_PAD   288    // 259 padded to 9*32

#define LDS_ROW_STRIDE 80          // 64B of K data + 16B pad -> conflict-free ds_load_b128
#define LDS_BUF_BYTES  (128 * LDS_ROW_STRIDE)   // 10240 B per buffer

typedef __bf16 bf16_t;
typedef __attribute__((ext_vector_type(16))) __bf16 v16bf;
typedef __attribute__((ext_vector_type(8)))  __bf16 v8bf;
typedef __attribute__((ext_vector_type(8)))  float  v8f;

union Frag16 { v16bf v; v8bf h[2]; };

// ---------------- CDNA5 async global->LDS helpers ----------------
// builtin signature (from hipcc diagnostic): first param is
// '__attribute__((vector_size(16))) int __device__ *' i.e. AS(1) v4i*, LDS arg AS(3) v4i*.
typedef int v4i_t __attribute__((vector_size(16)));
typedef __attribute__((address_space(1))) v4i_t* gv4i_p;
typedef __attribute__((address_space(3))) v4i_t* lv4i_p;

__device__ __forceinline__ void async_load_b128(const void* gsrc, void* lds) {
#if defined(__has_builtin) && __has_builtin(__builtin_amdgcn_global_load_async_to_lds_b128)
    v4i_t* g = (v4i_t*)gsrc;   // strip const, generic typed pointer
    v4i_t* l = (v4i_t*)lds;
    __builtin_amdgcn_global_load_async_to_lds_b128((gv4i_p)g, (lv4i_p)l, 0, 0);
#else
    unsigned loff = (unsigned)(uintptr_t)lds;
    asm volatile("global_load_async_to_lds_b128 %0, %1, off"
                 :: "v"(loff), "v"(gsrc) : "memory");
#endif
}

template <int N>
__device__ __forceinline__ void async_wait() {
#if defined(__has_builtin) && __has_builtin(__builtin_amdgcn_s_wait_asynccnt)
    __builtin_amdgcn_s_wait_asynccnt(N);
#else
    asm volatile("s_wait_asynccnt %0" :: "i"(N) : "memory");
#endif
}

// ---------------- helpers ----------------
__device__ __forceinline__ unsigned f2o(float f) {
    unsigned u = __float_as_uint(f);
    return (u & 0x80000000u) ? ~u : (u | 0x80000000u);
}
__device__ __forceinline__ float o2f(unsigned u) {
    return (u & 0x80000000u) ? __uint_as_float(u ^ 0x80000000u)
                             : __uint_as_float(~u);
}
__device__ __forceinline__ float gelu_exact(float v) {
    return 0.5f * v * (1.0f + erff(v * 0.70710678118654752f));
}

// ---------------- fills ----------------
__global__ void fill_f32(float* __restrict__ p, float v, size_t n) {
    size_t i = (size_t)blockIdx.x * blockDim.x + threadIdx.x;
    if (i < n) p[i] = v;
}
__global__ void fill_u32(unsigned* __restrict__ p, unsigned v, size_t n) {
    size_t i = (size_t)blockIdx.x * blockDim.x + threadIdx.x;
    if (i < n) p[i] = v;
}

// ---------------- fp32 -> bf16 activation conversion with K zero-pad ----------------
__global__ void cvt_act_f32_bf16(const float* __restrict__ src, bf16_t* __restrict__ dst,
                                 int M, int K, int Kpad) {
    size_t idx = (size_t)blockIdx.x * blockDim.x + threadIdx.x;
    size_t total = (size_t)M * Kpad;
    if (idx >= total) return;
    int k = (int)(idx % Kpad);
    size_t m = idx / Kpad;
    dst[idx] = (k < K) ? (bf16_t)src[m * (size_t)K + k] : (bf16_t)0.0f;
}

// gate input = [h (256) | het from x[:,384:387] | zeros] padded to 288, bf16
__global__ void build_gate_in(const float* __restrict__ h, const float* __restrict__ x,
                              bf16_t* __restrict__ dst) {
    size_t idx = (size_t)blockIdx.x * blockDim.x + threadIdx.x;
    size_t total = (size_t)N_NODES * KG_PAD;
    if (idx >= total) return;
    int k = (int)(idx % KG_PAD);
    size_t m = idx / KG_PAD;
    float v;
    if (k < HID)            v = h[m * HID + k];
    else if (k < HID + HET) v = x[m * IN_F + (IN_F - HET) + (k - HID)];
    else                    v = 0.0f;
    dst[idx] = (bf16_t)v;
}

// ---------------- pack fp32 weight [Kw][Nc] into WMMA B-fragment bf16 layout ----------------
// Bp[((nt*nKt + kt)*32 + lane)*16 + e]:
//   lane<16 : n = nt*16+lane,    k = kt*32 +     (e<8 ? e : e+8)   (K0..7,  K16..23)
//   lane>=16: n = nt*16+lane-16, k = kt*32 + 8 + (e<8 ? e : e+8)   (K8..15, K24..31)
__global__ void pack_wmma_b(const float* __restrict__ W, bf16_t* __restrict__ Bp,
                            int Kw, int Kpad, int Nc) {
    size_t idx = (size_t)blockIdx.x * blockDim.x + threadIdx.x;
    size_t total = (size_t)Kpad * Nc;
    if (idx >= total) return;
    int nKt = Kpad >> 5;
    int e    = (int)(idx & 15);
    int lane = (int)((idx >> 4) & 31);
    size_t t = idx >> 9;                 // nt*nKt + kt
    int kt = (int)(t % nKt);
    int nt = (int)(t / nKt);
    int k = (kt << 5) + ((lane >> 4) << 3) + (e < 8 ? e : e + 8);
    int n = (nt << 4) + (lane & 15);
    float v = (k < Kw) ? W[(size_t)k * Nc + n] : 0.0f;
    Bp[idx] = (bf16_t)v;
}

// ---------------- bf16 WMMA GEMM with async-LDS A staging ----------------
// C[M_PAD][Nc] = A[M_PAD][Kpad] @ Bp + bias
// Block: 256 thr = 8 waves; block tile 128Mx128N; wave tile 32Mx64N (8 accums).
// A tile (128x32 bf16) staged to LDS via global_load_async_to_lds_b128, double buffered.
__global__ void __launch_bounds__(256)
gemm_bf16_wmma(const bf16_t* __restrict__ A, const bf16_t* __restrict__ Bp,
               const float* __restrict__ bias, float* __restrict__ C,
               int Kpad, int Nc) {
    __shared__ __align__(16) char ldsA[2 * LDS_BUF_BYTES];

    const int nKt  = Kpad >> 5;
    const int tid  = threadIdx.x;
    const int lane = tid & 31;
    const int wid  = tid >> 5;
    const int mw   = wid >> 1;          // 0..3 : wave row group (32 rows each)
    const int nw   = wid & 1;           // 0..1 : wave col group (64 cols each)
    const int halfSel = lane >> 4;

    const int row0 = blockIdx.x * 128;                  // block row base
    const int n64  = blockIdx.y * 2 + nw;               // 64-col group index
    const size_t aBaseByte = (size_t)row0 * Kpad * 2;   // byte offset of block's A rows

    // staging map: q = tid*2 + i -> row = q>>2 (0..127), seg = q&3 (16B within 64B K-slab)
    const int q0 = tid * 2;

    auto stage = [&](int kt, int buf) {
#pragma unroll
        for (int i = 0; i < 2; ++i) {
            int q   = q0 + i;
            int row = q >> 2;
            int seg = q & 3;
            const char* g = (const char*)A + aBaseByte + (size_t)row * (Kpad * 2)
                            + kt * 64 + seg * 16;
            char* l = ldsA + buf * LDS_BUF_BYTES + row * LDS_ROW_STRIDE + seg * 16;
            async_load_b128(g, l);
        }
    };

    v8f c00 = {}, c01 = {}, c02 = {}, c03 = {};   // rows mw*32 + [0,16)
    v8f c10 = {}, c11 = {}, c12 = {}, c13 = {};   // rows mw*32 + [16,32)

    stage(0, 0);

    for (int kt = 0; kt < nKt; ++kt) {
        const int cur = kt & 1;
        if (kt + 1 < nKt) { stage(kt + 1, cur ^ 1); async_wait<2>(); }
        else              { async_wait<0>(); }
        __syncthreads();

        // A fragments from LDS (conflict-free: bank = row*20 mod 64)
        const char* bufp = ldsA + cur * LDS_BUF_BYTES;
        const char* pa0  = bufp + (mw * 32 + (lane & 15)) * LDS_ROW_STRIDE + halfSel * 16;
        const char* pa1  = pa0 + 16 * LDS_ROW_STRIDE;
        Frag16 a0, a1;
        a0.h[0] = *(const v8bf*)(pa0);
        a0.h[1] = *(const v8bf*)(pa0 + 32);
        a1.h[0] = *(const v8bf*)(pa1);
        a1.h[1] = *(const v8bf*)(pa1 + 32);

        Frag16 b;
#define DO_TILE(C0, C1, J)                                                          \
        {                                                                           \
            const bf16_t* bp = Bp +                                                 \
                (((((size_t)(n64 * 4 + (J)) * nKt + kt) << 5) + lane) << 4);        \
            b.h[0] = *(const v8bf*)(bp);                                            \
            b.h[1] = *(const v8bf*)(bp + 8);                                        \
            C0 = __builtin_amdgcn_wmma_f32_16x16x32_bf16(                           \
                     false, a0.v, false, b.v, (short)0, C0, false, false);          \
            C1 = __builtin_amdgcn_wmma_f32_16x16x32_bf16(                           \
                     false, a1.v, false, b.v, (short)0, C1, false, false);          \
        }
        DO_TILE(c00, c10, 0) DO_TILE(c01, c11, 1) DO_TILE(c02, c12, 2) DO_TILE(c03, c13, 3)
#undef DO_TILE
        __syncthreads();
    }

    // C/D layout: VGPR r -> M = r (lanes 0-15) / 8+r (lanes 16-31), N = lane&15
    const int col = (n64 << 6) + (lane & 15);
    const float b0 = bias[col], b1 = bias[col + 16], b2 = bias[col + 32], b3 = bias[col + 48];
    const int rbase0 = row0 + mw * 32 + halfSel * 8;
#pragma unroll
    for (int r = 0; r < 8; ++r) {
        size_t ro = (size_t)(rbase0 + r) * Nc;
        C[ro + col +  0] = c00[r] + b0;
        C[ro + col + 16] = c01[r] + b1;
        C[ro + col + 32] = c02[r] + b2;
        C[ro + col + 48] = c03[r] + b3;
        ro += (size_t)16 * Nc;
        C[ro + col +  0] = c10[r] + b0;
        C[ro + col + 16] = c11[r] + b1;
        C[ro + col + 32] = c12[r] + b2;
        C[ro + col + 48] = c13[r] + b3;
    }
}

// ---------------- edge score: wave per edge ----------------
__global__ void __launch_bounds__(256)
edge_score(const int* __restrict__ ei, const float* __restrict__ xl,
           const float* __restrict__ xr, const float* __restrict__ att,
           float* __restrict__ score) {
    int w    = (int)((blockIdx.x * blockDim.x + threadIdx.x) >> 5);
    int lane = threadIdx.x & 31;
    if (w >= E2) return;
    int s, d;
    if (w < E_EDGES) { s = ei[w]; d = ei[E_EDGES + w]; } else { s = d = w - E_EDGES; }

    const float4* ps = (const float4*)(xl + (size_t)s * HID + lane * 8);
    const float4* pd = (const float4*)(xr + (size_t)d * HID + lane * 8);
    const float4* pa = (const float4*)(att + lane * 8);
    float acc = 0.0f;
#pragma unroll
    for (int i = 0; i < 2; ++i) {
        float4 a4 = ps[i], d4 = pd[i], t4 = pa[i];
        float v;
        v = a4.x + d4.x; v = v > 0.f ? v : 0.2f * v; acc += v * t4.x;
        v = a4.y + d4.y; v = v > 0.f ? v : 0.2f * v; acc += v * t4.y;
        v = a4.z + d4.z; v = v > 0.f ? v : 0.2f * v; acc += v * t4.z;
        v = a4.w + d4.w; v = v > 0.f ? v : 0.2f * v; acc += v * t4.w;
    }
    acc += __shfl_xor(acc, 1);
    acc += __shfl_xor(acc, 2);
    acc += __shfl_xor(acc, 4);
    if ((lane & 7) == 0) score[(size_t)w * HEADS + (lane >> 3)] = acc;
}

// ---------------- segment softmax over dst (per head) ----------------
__global__ void seg_reduce_max(const int* __restrict__ ei, const float* __restrict__ score,
                               unsigned* __restrict__ segmaxU) {
    size_t idx = (size_t)blockIdx.x * blockDim.x + threadIdx.x;
    if (idx >= (size_t)E2 * HEADS) return;
    int e  = (int)(idx >> 2);
    int hh = (int)(idx & 3);
    int d  = (e < E_EDGES) ? ei[E_EDGES + e] : (e - E_EDGES);
    atomicMax(&segmaxU[d * HEADS + hh], f2o(score[idx]));
}

__global__ void seg_exp_sum(const int* __restrict__ ei, const float* __restrict__ score,
                            const unsigned* __restrict__ segmaxU, float* __restrict__ segsum) {
    size_t idx = (size_t)blockIdx.x * blockDim.x + threadIdx.x;
    if (idx >= (size_t)E2 * HEADS) return;
    int e  = (int)(idx >> 2);
    int hh = (int)(idx & 3);
    int d  = (e < E_EDGES) ? ei[E_EDGES + e] : (e - E_EDGES);
    float m = o2f(segmaxU[d * HEADS + hh]);
    atomicAdd(&segsum[d * HEADS + hh], expf(score[idx] - m));
}

// ---------------- message scatter: out[dst] += alpha * xl[src] ----------------
__global__ void __launch_bounds__(256)
edge_message(const int* __restrict__ ei, const float* __restrict__ score,
             const unsigned* __restrict__ segmaxU, const float* __restrict__ segsum,
             const float* __restrict__ xl, float* __restrict__ outacc) {
    int w    = (int)((blockIdx.x * blockDim.x + threadIdx.x) >> 5);
    int lane = threadIdx.x & 31;
    if (w >= E2) return;
    int s, d;
    if (w < E_EDGES) { s = ei[w]; d = ei[E_EDGES + w]; } else { s = d = w - E_EDGES; }
    int hh = lane >> 3;
    float m     = o2f(segmaxU[d * HEADS + hh]);
    float alpha = expf(score[(size_t)w * HEADS + hh] - m) / segsum[d * HEADS + hh];

    const float4* ps = (const float4*)(xl + (size_t)s * HID + lane * 8);
    float*        po = outacc + (size_t)d * HID + lane * 8;
    float4 v0 = ps[0], v1 = ps[1];
    atomicAdd(po + 0, alpha * v0.x);
    atomicAdd(po + 1, alpha * v0.y);
    atomicAdd(po + 2, alpha * v0.z);
    atomicAdd(po + 3, alpha * v0.w);
    atomicAdd(po + 4, alpha * v1.x);
    atomicAdd(po + 5, alpha * v1.y);
    atomicAdd(po + 6, alpha * v1.z);
    atomicAdd(po + 7, alpha * v1.w);
}

// ---------------- LayerNorm (+optional bias, residual, GELU) : wave per node ----------------
__global__ void __launch_bounds__(256)
ln_kernel(const float* __restrict__ in, const float* __restrict__ bias,
          const float* __restrict__ resid, const float* __restrict__ g,
          const float* __restrict__ bta, float* __restrict__ outH, int doGelu) {
    int node = (int)((blockIdx.x * blockDim.x + threadIdx.x) >> 5);
    int lane = threadIdx.x & 31;
    if (node >= N_NODES) return;
    size_t base = (size_t)node * HID + lane * 8;
    float v[8];
#pragma unroll
    for (int i = 0; i < 8; ++i) {
        float t = in[base + i];
        if (bias)  t += bias[lane * 8 + i];
        if (resid) t += resid[base + i];
        v[i] = t;
    }
    float s = 0.f, s2 = 0.f;
#pragma unroll
    for (int i = 0; i < 8; ++i) { s += v[i]; s2 += v[i] * v[i]; }
#pragma unroll
    for (int m = 1; m < 32; m <<= 1) { s += __shfl_xor(s, m); s2 += __shfl_xor(s2, m); }
    float mu  = s * (1.0f / HID);
    float var = s2 * (1.0f / HID) - mu * mu;
    float inv = rsqrtf(var + 1e-5f);
#pragma unroll
    for (int i = 0; i < 8; ++i) {
        float y = (v[i] - mu) * inv * g[lane * 8 + i] + bta[lane * 8 + i];
        if (doGelu) y = gelu_exact(y);
        outH[base + i] = y;
    }
}

// ---------------- gate: raw[i] = tanh(gt[i,:]) . W2 + b2 : wave per node ----------------
__global__ void __launch_bounds__(256)
gate_raw_kernel(const float* __restrict__ gt, const float* __restrict__ W2,
                const float* __restrict__ b2, float* __restrict__ raw) {
    int node = (int)((blockIdx.x * blockDim.x + threadIdx.x) >> 5);
    int lane = threadIdx.x & 31;
    if (node >= N_NODES) return;
    float acc = 0.f;
#pragma unroll
    for (int i = 0; i < 4; ++i) {
        int j = lane * 4 + i;
        acc += tanhf(gt[(size_t)node * AH + j]) * W2[j];
    }
#pragma unroll
    for (int m = 1; m < 32; m <<= 1) acc += __shfl_xor(acc, m);
    if (lane == 0) raw[node] = acc + b2[0];
}

// ---------------- batch segment softmax (B=8) ----------------
__global__ void batch_max_kernel(const int* __restrict__ batch, const float* __restrict__ raw,
                                 unsigned* __restrict__ bmaxU) {
    int i = (int)(blockIdx.x * blockDim.x + threadIdx.x);
    if (i >= N_NODES) return;
    atomicMax(&bmaxU[batch[i]], f2o(raw[i]));
}
__global__ void batch_expsum_kernel(const int* __restrict__ batch, const float* __restrict__ raw,
                                    const unsigned* __restrict__ bmaxU, float* __restrict__ bsum) {
    int i = (int)(blockIdx.x * blockDim.x + threadIdx.x);
    if (i >= N_NODES) return;
    int b = batch[i];
    atomicAdd(&bsum[b], expf(raw[i] - o2f(bmaxU[b])));
}

// ---------------- pooled z[b,:] = sum attn[i]*values[i,:]; batch sorted -> run-flush ----------------
__global__ void __launch_bounds__(256)
z_reduce(const int* __restrict__ batch, const float* __restrict__ raw,
         const unsigned* __restrict__ bmaxU, const float* __restrict__ bsum,
         const float* __restrict__ values, float* __restrict__ z) {
    int d  = threadIdx.x;            // 0..255 = feature dim
    int n0 = blockIdx.x * 64;
    if (n0 >= N_NODES) return;
    int cur = batch[n0];
    float acc = 0.f;
    for (int i = 0; i < 64; ++i) {
        int n = n0 + i;
        if (n >= N_NODES) break;
        int bb = batch[n];
        if (bb != cur) { atomicAdd(&z[cur * HID + d], acc); acc = 0.f; cur = bb; }
        float a = expf(raw[n] - o2f(bmaxU[bb])) / bsum[bb];
        acc += a * values[(size_t)n * HID + d];
    }
    atomicAdd(&z[cur * HID + d], acc);
}

// ---------------- head MLP: logits[b] = gelu(z@W1+b1) @ W2 + b2 ----------------
__global__ void head_kernel(const float* __restrict__ z, const float* __restrict__ W1,
                            const float* __restrict__ b1, const float* __restrict__ W2,
                            const float* __restrict__ b2, float* __restrict__ logits) {
    __shared__ float red[AH];
    int j = threadIdx.x;   // 0..127
    for (int b = 0; b < B_GRAPHS; ++b) {
        float acc = b1[j];
        for (int k = 0; k < HID; ++k) acc += z[b * HID + k] * W1[(size_t)k * AH + j];
        red[j] = gelu_exact(acc) * W2[j];
        __syncthreads();
        for (int st = 64; st > 0; st >>= 1) {
            if (j < st) red[j] += red[j + st];
            __syncthreads();
        }
        if (j == 0) logits[b] = red[0] + b2[0];
        __syncthreads();
    }
}

// ================= host launch =================
extern "C" void kernel_launch(void* const* d_in, const int* in_sizes, int n_in,
                              void* d_out, int out_size, void* d_ws, size_t ws_size,
                              hipStream_t stream) {
    (void)in_sizes; (void)n_in; (void)out_size; (void)ws_size;

    const float* x      = (const float*)d_in[0];
    const int*   ei     = (const int*)d_in[1];
    const int*   batch  = (const int*)d_in[2];
    const float* in_W   = (const float*)d_in[3];
    const float* in_b   = (const float*)d_in[4];
    const float* in_g   = (const float*)d_in[5];
    const float* in_bt  = (const float*)d_in[6];
    const float* gate_W1 = (const float*)d_in[31];
    const float* gate_b1 = (const float*)d_in[32];
    const float* gate_W2 = (const float*)d_in[33];
    const float* gate_b2 = (const float*)d_in[34];
    const float* val_W   = (const float*)d_in[35];
    const float* val_b   = (const float*)d_in[36];
    const float* head_W1 = (const float*)d_in[37];
    const float* head_b1 = (const float*)d_in[38];
    const float* head_W2 = (const float*)d_in[39];
    const float* head_b2 = (const float*)d_in[40];
    float* logits = (float*)d_out;

    // ---- workspace carve-up (256B aligned); GEMM in/out buffers sized for M_PAD ----
    char* ws = (char*)d_ws;
    size_t off = 0;
    auto carve = [&](size_t bytes) -> char* {
        char* p = ws + off;
        off += (bytes + 255) & ~(size_t)255;
        return p;
    };
    float*    h       = (float*)carve((size_t)M_PAD * HID * 4);
    float*    tmp     = (float*)carve((size_t)M_PAD * HID * 4);   // GEMM out / msg accumulator
    float*    xl      = (float*)carve((size_t)M_PAD * HID * 4);   // also gate hidden (N x 128)
    float*    xr      = (float*)carve((size_t)M_PAD * HID * 4);   // also values
    bf16_t*   abuf    = (bf16_t*)carve((size_t)M_PAD * KIN_PAD * 2);
    float*    score   = (float*)carve((size_t)E2 * HEADS * 4);
    unsigned* segmaxU = (unsigned*)carve((size_t)N_NODES * HEADS * 4);
    float*    segsum  = (float*)carve((size_t)N_NODES * HEADS * 4);
    float*    raw     = (float*)carve((size_t)N_NODES * 4);
    unsigned* bmaxU   = (unsigned*)carve(64);
    float*    bsum    = (float*)carve(64);
    float*    z       = (float*)carve((size_t)B_GRAPHS * HID * 4);
    bf16_t*   wInP    = (bf16_t*)carve((size_t)KIN_PAD * HID * 2);
    bf16_t*   wLP[3], *wRP[3];
    for (int l = 0; l < 3; ++l) {
        wLP[l] = (bf16_t*)carve((size_t)HID * HID * 2);
        wRP[l] = (bf16_t*)carve((size_t)HID * HID * 2);
    }
    bf16_t*   wValP   = (bf16_t*)carve((size_t)KG_PAD * HID * 2);  // val_W packed w/ Kpad=288
    bf16_t*   wGateP  = (bf16_t*)carve((size_t)KG_PAD * AH * 2);

    auto blks = [](size_t total, int per) { return (unsigned)((total + per - 1) / per); };
    auto gemm = [&](const bf16_t* A, const bf16_t* Bp, const float* bias, float* C,
                    int Kpad, int Nc) {
        dim3 grid(M_PAD / 128, Nc / 128);
        gemm_bf16_wmma<<<grid, 256, 0, stream>>>(A, Bp, bias, C, Kpad, Nc);
    };

    // ---- pack weights into WMMA B-fragment layout (cheap, once per call) ----
    pack_wmma_b<<<blks((size_t)KIN_PAD * HID, 256), 256, 0, stream>>>(in_W, wInP, IN_F, KIN_PAD, HID);
    for (int l = 0; l < 3; ++l) {
        const float* Wl = (const float*)d_in[7 + 8 * l + 0];
        const float* Wr = (const float*)d_in[7 + 8 * l + 2];
        pack_wmma_b<<<blks((size_t)HID * HID, 256), 256, 0, stream>>>(Wl, wLP[l], HID, HID, HID);
        pack_wmma_b<<<blks((size_t)HID * HID, 256), 256, 0, stream>>>(Wr, wRP[l], HID, HID, HID);
    }
    pack_wmma_b<<<blks((size_t)KG_PAD * HID, 256), 256, 0, stream>>>(val_W, wValP, HID, KG_PAD, HID);
    pack_wmma_b<<<blks((size_t)KG_PAD * AH, 256), 256, 0, stream>>>(gate_W1, wGateP, HID + HET, KG_PAD, AH);

    // ---- input projection: h = gelu(LN(x @ in_W + in_b)) ----
    cvt_act_f32_bf16<<<blks((size_t)N_NODES * KIN_PAD, 256), 256, 0, stream>>>(x, abuf, N_NODES, IN_F, KIN_PAD);
    gemm(abuf, wInP, in_b, tmp, KIN_PAD, HID);
    ln_kernel<<<blks((size_t)N_NODES * 32, 256), 256, 0, stream>>>(tmp, nullptr, nullptr, in_g, in_bt, h, 1);

    // ---- 3 GATv2 layers ----
    for (int l = 0; l < 3; ++l) {
        const float* bl   = (const float*)d_in[7 + 8 * l + 1];
        const float* br   = (const float*)d_in[7 + 8 * l + 3];
        const float* att  = (const float*)d_in[7 + 8 * l + 4];
        const float* bias = (const float*)d_in[7 + 8 * l + 5];
        const float* lng  = (const float*)d_in[7 + 8 * l + 6];
        const float* lnb  = (const float*)d_in[7 + 8 * l + 7];

        cvt_act_f32_bf16<<<blks((size_t)N_NODES * HID, 256), 256, 0, stream>>>(h, abuf, N_NODES, HID, HID);
        gemm(abuf, wLP[l], bl, xl, HID, HID);
        gemm(abuf, wRP[l], br, xr, HID, HID);

        edge_score<<<blks((size_t)E2 * 32, 256), 256, 0, stream>>>(ei, xl, xr, att, score);

        fill_u32<<<blks((size_t)N_NODES * HEADS, 256), 256, 0, stream>>>(segmaxU, 0u, (size_t)N_NODES * HEADS);
        fill_f32<<<blks((size_t)N_NODES * HEADS, 256), 256, 0, stream>>>(segsum, 0.f, (size_t)N_NODES * HEADS);
        fill_f32<<<blks((size_t)N_NODES * HID, 256), 256, 0, stream>>>(tmp, 0.f, (size_t)N_NODES * HID);

        seg_reduce_max<<<blks((size_t)E2 * HEADS, 256), 256, 0, stream>>>(ei, score, segmaxU);
        seg_exp_sum<<<blks((size_t)E2 * HEADS, 256), 256, 0, stream>>>(ei, score, segmaxU, segsum);
        edge_message<<<blks((size_t)E2 * 32, 256), 256, 0, stream>>>(ei, score, segmaxU, segsum, xl, tmp);

        // out = msg + bias (+ h residual for l>0); h = LN(out)
        ln_kernel<<<blks((size_t)N_NODES * 32, 256), 256, 0, stream>>>(
            tmp, bias, (l > 0) ? h : nullptr, lng, lnb, h, 0);
    }

    // ---- readout ----
    // gate input [h|het|0] bf16 at K=288 feeds BOTH gate GEMM and value GEMM
    build_gate_in<<<blks((size_t)N_NODES * KG_PAD, 256), 256, 0, stream>>>(h, x, abuf);
    gemm(abuf, wGateP, gate_b1, xl, KG_PAD, AH);
    gemm(abuf, wValP, val_b, xr, KG_PAD, HID);

    gate_raw_kernel<<<blks((size_t)N_NODES * 32, 256), 256, 0, stream>>>(xl, gate_W2, gate_b2, raw);

    fill_u32<<<1, 32, 0, stream>>>(bmaxU, 0u, B_GRAPHS);
    fill_f32<<<1, 32, 0, stream>>>(bsum, 0.f, B_GRAPHS);
    fill_f32<<<blks((size_t)B_GRAPHS * HID, 256), 256, 0, stream>>>(z, 0.f, (size_t)B_GRAPHS * HID);

    batch_max_kernel<<<blks(N_NODES, 256), 256, 0, stream>>>(batch, raw, bmaxU);
    batch_expsum_kernel<<<blks(N_NODES, 256), 256, 0, stream>>>(batch, raw, bmaxU, bsum);
    z_reduce<<<blks((size_t)N_NODES, 64), 256, 0, stream>>>(batch, raw, bmaxU, bsum, xr, z);

    head_kernel<<<1, AH, 0, stream>>>(z, head_W1, head_b1, head_W2, head_b2, logits);
}
